// NodeAdder_21801253994886
// MI455X (gfx1250) — compile-verified, matrix operand order checked by reading, not run on hardware
//
#include <hip/hip_runtime.h>
#include <hip/hip_bf16.h>
#include <stdint.h>

// Problem constants (match reference)
#define Ssz   128        // state size
#define Asz   256        // aggregated size
#define Bsz   128        // batch
#define Nn    65536      // nodes
#define Ee    655360     // edges
#define NSTEP 2
#define Tt    32         // node types
#define Msz   256        // message size = 2S
#define G3S   384        // 3S

typedef __attribute__((ext_vector_type(16))) __bf16 v16bf;
typedef __attribute__((ext_vector_type(8)))  float  v8f;

#define WMMA_BF16(a, b, c) \
  __builtin_amdgcn_wmma_f32_16x16x32_bf16(false, (a), false, (b), (short)0, (c), false, false)

// Load a 16x32 bf16 WMMA fragment from row-major [rows, K] at (row, k0).
// Per ISA layout: lanes 0-15 hold K = k0+0..7 and k0+16..23; lanes 16-31 hold
// K = k0+8..15 and k0+24..31 (caller passes p already offset by hi*8).
// p must be 16-byte aligned (K multiple of 32 floats, offsets multiple of 8).
__device__ __forceinline__ v16bf load_frag_bf16(const float* __restrict__ p) {
  const float4* p4 = (const float4*)p;
  const float4 x0 = p4[0];   // K offsets +0..3
  const float4 x1 = p4[1];   // +4..7
  const float4 y0 = p4[4];   // +16..19
  const float4 y1 = p4[5];   // +20..23
  v16bf f;
  f[0]  = (__bf16)x0.x; f[1]  = (__bf16)x0.y; f[2]  = (__bf16)x0.z; f[3]  = (__bf16)x0.w;
  f[4]  = (__bf16)x1.x; f[5]  = (__bf16)x1.y; f[6]  = (__bf16)x1.z; f[7]  = (__bf16)x1.w;
  f[8]  = (__bf16)y0.x; f[9]  = (__bf16)y0.y; f[10] = (__bf16)y0.z; f[11] = (__bf16)y0.w;
  f[12] = (__bf16)y1.x; f[13] = (__bf16)y1.y; f[14] = (__bf16)y1.z; f[15] = (__bf16)y1.w;
  return f;
}

// ---------------------------------------------------------------------------
// Generic NT GEMM on WMMA:  C[R,O] = (accum? C:0) + A[R,K] @ W[O,K]^T + bias[O]
// A, W row-major (K contiguous) -> both fragments load identically.
// Register-blocked: one wave owns a 32x32 C macro-tile (4 accumulators,
// 2 A-frags x 2 B-frags -> 4 WMMAs per K step; ~8 FLOP/byte from cache).
// 4 waves (128 threads) per block over col macro-tiles.
// R multiple of 32; K multiple of 32; O arbitrary (OOB cols: clamped B row,
// garbage confined to masked columns -> K loop stays branch-free, EXEC=all-1).
// ---------------------------------------------------------------------------
__global__ void gemm_wmma_nt(const float* __restrict__ A, const float* __restrict__ W,
                             const float* __restrict__ bias, float* __restrict__ C,
                             int R, int K, int O, int accum) {
  const int lane = threadIdx.x & 31;
  const int wave = threadIdx.x >> 5;
  const int lr   = lane & 15;        // row within tile (A) / col within tile (B)
  const int hi   = lane >> 4;        // half-wave selects K sub-bands
  const int o0   = (blockIdx.x * 4 + wave) * 32;
  if (o0 >= O) return;               // wave-uniform: EXEC all-ones for WMMA
  const int r0   = blockIdx.y * 32;

  int oc0 = o0 + lr;      const bool ov0 = (oc0 < O); if (!ov0) oc0 = O - 1;
  int oc1 = o0 + 16 + lr; const bool ov1 = (oc1 < O); if (!ov1) oc1 = O - 1;

  const float* a0p = A + (size_t)(r0 + lr) * K + hi * 8;
  const float* a1p = A + (size_t)(r0 + 16 + lr) * K + hi * 8;
  const float* b0p = W + (size_t)oc0 * K + hi * 8;
  const float* b1p = W + (size_t)oc1 * K + hi * 8;

  v8f c00 = {}, c01 = {}, c10 = {}, c11 = {};
  for (int k0 = 0; k0 < K; k0 += 32) {
    const v16bf a0 = load_frag_bf16(a0p + k0);
    const v16bf a1 = load_frag_bf16(a1p + k0);
    const v16bf b0 = load_frag_bf16(b0p + k0);
    const v16bf b1 = load_frag_bf16(b1p + k0);
    c00 = WMMA_BF16(a0, b0, c00);
    c01 = WMMA_BF16(a0, b1, c01);
    c10 = WMMA_BF16(a1, b0, c10);
    c11 = WMMA_BF16(a1, b1, c11);
  }

  const float bb0 = bias ? bias[oc0] : 0.0f;   // oc clamped -> always in-bounds
  const float bb1 = bias ? bias[oc1] : 0.0f;
  // C layout: VGPR g -> row (rbase + g + hi*8), col = lr part of oc
#pragma unroll
  for (int q = 0; q < 4; ++q) {
    const v8f&  acc = (q == 0) ? c00 : (q == 1) ? c01 : (q == 2) ? c10 : c11;
    const int   rb  = r0 + ((q >> 1) ? 16 : 0);
    const int   oc  = (q & 1) ? oc1 : oc0;
    const bool  ov  = (q & 1) ? ov1 : ov0;
    const float bb  = (q & 1) ? bb1 : bb0;
    if (!ov) continue;               // divergence only affects stores
    if (accum) {
#pragma unroll
      for (int g = 0; g < 8; ++g) {
        const size_t idx = (size_t)(rb + g + hi * 8) * O + oc;
        C[idx] = C[idx] + acc[g] + bb;
      }
    } else {
#pragma unroll
      for (int g = 0; g < 8; ++g) {
        const size_t idx = (size_t)(rb + g + hi * 8) * O + oc;
        C[idx] = acc[g] + bb;
      }
    }
  }
}

// ---------------------------------------------------------------------------
// Fused edge-message kernel (the dominant GEMM + gather + segment-sum):
//   feat_t = edge_features @ Wfeat^T        (WMMA, never materialized)
//   msg    = feat_t + src_t[esrc] + dest_t[edst]   (gathers hit 67MB L2 tables)
//   agg[edst] += msg                        (f32 global atomics, L2-resident)
// Block = 256 threads (8 waves) covers 16 edges x all 256 message cols:
// wave w owns col tiles w and w+8 (A fragment reused across both).
// ---------------------------------------------------------------------------
__global__ void edge_msg_wmma(const float* __restrict__ EF,    // [E,S]
                              const float* __restrict__ Wf,    // [M,S]
                              const float* __restrict__ src_t, // [N,M]
                              const float* __restrict__ dst_t, // [N,M]
                              const int*   __restrict__ esrc,
                              const int*   __restrict__ edst,
                              float*       __restrict__ agg) { // [N,M]
  const int lane = threadIdx.x & 31;
  const int wave = threadIdx.x >> 5;   // 0..7
  const int lr   = lane & 15;
  const int hi   = lane >> 4;
  const int e0   = blockIdx.x * 16;
  const int oc0  = wave * 16;
  const int oc1  = (wave + 8) * 16;

  const float* arow  = EF + (size_t)(e0 + lr) * Ssz + hi * 8;
  const float* brow0 = Wf + (size_t)(oc0 + lr) * Ssz + hi * 8;
  const float* brow1 = Wf + (size_t)(oc1 + lr) * Ssz + hi * 8;

  v8f acc0 = {}, acc1 = {};
#pragma unroll
  for (int k0 = 0; k0 < Ssz; k0 += 32) {
    const v16bf a  = load_frag_bf16(arow + k0);
    const v16bf b0 = load_frag_bf16(brow0 + k0);
    const v16bf b1 = load_frag_bf16(brow1 + k0);
    acc0 = WMMA_BF16(a, b0, acc0);
    acc1 = WMMA_BF16(a, b1, acc1);
  }

  // C-tile row g maps to edge e0 + g + hi*8: 8 consecutive edge ids per lane.
  const int4* sp = (const int4*)(esrc + e0 + hi * 8);
  const int4* dp = (const int4*)(edst + e0 + hi * 8);
  const int4 s0 = sp[0], s1 = sp[1];
  const int4 d0 = dp[0], d1 = dp[1];
  const int sidx[8] = {s0.x, s0.y, s0.z, s0.w, s1.x, s1.y, s1.z, s1.w};
  const int didx[8] = {d0.x, d0.y, d0.z, d0.w, d1.x, d1.y, d1.z, d1.w};

#pragma unroll
  for (int g = 0; g < 8; ++g) {
    const size_t so = (size_t)sidx[g] * Msz;
    const size_t dd = (size_t)didx[g] * Msz;
    const int c0 = oc0 + lr;
    const int c1 = oc1 + lr;
    atomicAdd(agg + dd + c0, acc0[g] + src_t[so + c0] + dst_t[dd + c0]);
    atomicAdd(agg + dd + c1, acc1[g] + src_t[so + c1] + dst_t[dd + c1]);
  }
}

// ---------------------------------------------------------------------------
// GRUCell pointwise: h = (1-z)*n + z*h   (all nodes update: owner mask all-true)
// ---------------------------------------------------------------------------
__global__ void gru_update(const float* __restrict__ gi, const float* __restrict__ gh,
                           float* __restrict__ h) {
  const size_t idx = (size_t)blockIdx.x * blockDim.x + threadIdx.x;
  if (idx >= (size_t)Nn * Ssz) return;
  const size_t n = idx / Ssz, s = idx % Ssz;
  const size_t b = n * G3S;
  const float ir  = gi[b + s], iz = gi[b + Ssz + s], in_ = gi[b + 2 * Ssz + s];
  const float hr  = gh[b + s], hz = gh[b + Ssz + s], hn  = gh[b + 2 * Ssz + s];
  const float r   = 1.0f / (1.0f + __expf(-(ir + hr)));
  const float z   = 1.0f / (1.0f + __expf(-(iz + hz)));
  const float nn  = tanhf(in_ + r * hn);
  const size_t hp = n * Ssz + s;
  h[hp] = (1.0f - z) * nn + z * h[hp];
}

// out[owner[n], a] += data[n,a] * sigmoid(gates[n,a])
__global__ void agg_scatter(const float* __restrict__ gates_pre,
                            const float* __restrict__ data_pre,
                            const int*   __restrict__ owner,
                            float*       __restrict__ out) {   // [B,A]
  const size_t idx = (size_t)blockIdx.x * blockDim.x + threadIdx.x;
  if (idx >= (size_t)Nn * Asz) return;
  const size_t n = idx / Asz, a = idx % Asz;
  const float g = 1.0f / (1.0f + __expf(-gates_pre[idx]));
  atomicAdd(out + (size_t)owner[n] * Asz + a, data_pre[idx] * g);
}

__global__ void zero_f32(float* __restrict__ p, long long n) {
  long long i = (long long)blockIdx.x * blockDim.x + threadIdx.x;
  const long long stride = (long long)gridDim.x * blockDim.x;
  for (; i < n; i += stride) p[i] = 0.0f;
}

// cross-entropy over 33 logits; one block of 128 threads (one row each)
__global__ void loss_kernel(const float* __restrict__ logits,   // [B, T+1]
                            const int*   __restrict__ ref_types,
                            float*       __restrict__ loss_out) {
  __shared__ float red[Bsz];
  const int b = threadIdx.x;
  const float* L = logits + b * (Tt + 1);
  float mx = L[0];
  for (int j = 1; j < Tt + 1; ++j) mx = fmaxf(mx, L[j]);
  float se = 0.0f;
  for (int j = 0; j < Tt + 1; ++j) se += __expf(L[j] - mx);
  const int t = ref_types[b] + 1;          // remap_pad
  red[b] = -(L[t] - mx - __logf(se));
  __syncthreads();
  for (int off = Bsz / 2; off > 0; off >>= 1) {
    if (b < off) red[b] += red[b + off];
    __syncthreads();
  }
  if (b == 0) *loss_out = red[0] / (float)Bsz;
}

__global__ void emb_gather(const float* __restrict__ type_emb,
                           const int* __restrict__ ref_types,
                           float* __restrict__ emb) {           // [B,S]
  const int idx = blockIdx.x * blockDim.x + threadIdx.x;
  if (idx >= Bsz * Ssz) return;
  const int b = idx / Ssz, s = idx % Ssz;
  emb[idx] = type_emb[ref_types[b] * Ssz + s];
}

// ---------------------------------------------------------------------------
static void launch_gemm(const float* A, const float* W, const float* bias, float* C,
                        int R, int K, int O, int accum, hipStream_t s) {
  const int otiles = (O + 31) / 32;          // 32-wide col macro-tiles
  dim3 grid((otiles + 3) / 4, R / 32);       // R must be a multiple of 32
  gemm_wmma_nt<<<grid, 128, 0, s>>>(A, W, bias, C, R, K, O, accum);
}

static void launch_zero(float* p, long long n, hipStream_t s) {
  zero_f32<<<4096, 256, 0, s>>>(p, n);
}

extern "C" void kernel_launch(void* const* d_in, const int* in_sizes, int n_in,
                              void* d_out, int out_size, void* d_ws, size_t ws_size,
                              hipStream_t stream) {
  const float* nodes_in = (const float*)d_in[0];
  const float* edge_fts = (const float*)d_in[1];
  const float* Wsrc  = (const float*)d_in[2];
  const float* Wdest = (const float*)d_in[3];
  const float* bdest = (const float*)d_in[4];
  const float* Wfeat = (const float*)d_in[5];
  const float* Wih   = (const float*)d_in[6];
  const float* Whh   = (const float*)d_in[7];
  const float* bih   = (const float*)d_in[8];
  const float* bhh   = (const float*)d_in[9];
  const float* dWt   = (const float*)d_in[10];
  const float* dbt   = (const float*)d_in[11];
  const float* dWg   = (const float*)d_in[12];
  const float* dbg   = (const float*)d_in[13];
  const float* iWt   = (const float*)d_in[14];
  const float* ibt   = (const float*)d_in[15];
  const float* iWg   = (const float*)d_in[16];
  const float* ibg   = (const float*)d_in[17];
  const float* decW  = (const float*)d_in[18];
  const float* decb  = (const float*)d_in[19];
  const float* temb  = (const float*)d_in[20];
  const float* f1W   = (const float*)d_in[21];
  const float* f1b   = (const float*)d_in[22];
  const float* f2W   = (const float*)d_in[23];
  const int* esrc    = (const int*)d_in[24];
  const int* edst    = (const int*)d_in[25];
  const int* rtypes  = (const int*)d_in[26];
  const int* owner   = (const int*)d_in[27];

  // Workspace carve-up (~302 MB): nodes + two N*3S scratch mats + agg + smalls
  char* ws = (char*)d_ws;
  float* nodes = (float*)ws; ws += (size_t)Nn * Ssz * 4;   // 33.5 MB (mutable h)
  float* bufA  = (float*)ws; ws += (size_t)Nn * G3S * 4;   // 100.7 MB
  float* bufB  = (float*)ws; ws += (size_t)Nn * G3S * 4;   // 100.7 MB
  float* aggm  = (float*)ws; ws += (size_t)Nn * Msz * 4;   // 67.1 MB
  float* aggdec  = (float*)ws; ws += (size_t)Bsz * Asz * 4;
  float* agginit = (float*)ws; ws += (size_t)Bsz * Asz * 4;
  float* logits  = (float*)ws; ws += (size_t)Bsz * (Tt + 1) * 4;
  float* embB    = (float*)ws; ws += (size_t)Bsz * Ssz * 4;
  float* newfeat = (float*)ws; ws += (size_t)Bsz * Ssz * 4;

  float* out = (float*)d_out;
  float* loss_slot = out + (size_t)(Nn + Bsz) * Ssz;

  hipMemcpyAsync(nodes, nodes_in, (size_t)Nn * Ssz * 4,
                 hipMemcpyDeviceToDevice, stream);

  for (int i = 0; i < NSTEP; ++i) {
    // src_t / dest_t (bufA/bufB reuse first N*M floats)
    launch_gemm(nodes, Wsrc  + (size_t)i * Msz * Ssz, nullptr,          bufA, Nn, Ssz, Msz, 0, stream);
    launch_gemm(nodes, Wdest + (size_t)i * Msz * Ssz, bdest + i * Msz, bufB, Nn, Ssz, Msz, 0, stream);
    launch_zero(aggm, (long long)Nn * Msz, stream);
    edge_msg_wmma<<<Ee / 16, 256, 0, stream>>>(edge_fts,
        Wfeat + (size_t)i * Msz * Ssz, bufA, bufB, esrc, edst, aggm);
    // GRU gates: gi = agg @ Wih^T + bih ; gh = h @ Whh^T + bhh
    launch_gemm(aggm,  Wih + (size_t)i * G3S * Msz, bih + i * G3S, bufA, Nn, Msz, G3S, 0, stream);
    launch_gemm(nodes, Whh + (size_t)i * G3S * Ssz, bhh + i * G3S, bufB, Nn, Ssz, G3S, 0, stream);
    gru_update<<<((size_t)Nn * Ssz + 255) / 256, 256, 0, stream>>>(bufA, bufB, nodes);
  }

  // decision aggregator
  launch_gemm(nodes, dWg, dbg, bufA, Nn, Ssz, Asz, 0, stream);   // gates pre-act
  launch_gemm(nodes, dWt, dbt, bufB, Nn, Ssz, Asz, 0, stream);   // data
  launch_zero(aggdec, (long long)Bsz * Asz, stream);
  agg_scatter<<<((size_t)Nn * Asz + 255) / 256, 256, 0, stream>>>(bufA, bufB, owner, aggdec);

  // init aggregator
  launch_gemm(nodes, iWg, ibg, bufA, Nn, Ssz, Asz, 0, stream);
  launch_gemm(nodes, iWt, ibt, bufB, Nn, Ssz, Asz, 0, stream);
  launch_zero(agginit, (long long)Bsz * Asz, stream);
  agg_scatter<<<((size_t)Nn * Asz + 255) / 256, 256, 0, stream>>>(bufA, bufB, owner, agginit);

  // logits -> loss
  launch_gemm(aggdec, decW, decb, logits, Bsz, Asz, Tt + 1, 0, stream);
  loss_kernel<<<1, Bsz, 0, stream>>>(logits, rtypes, loss_slot);

  // new node features
  emb_gather<<<(Bsz * Ssz + 255) / 256, 256, 0, stream>>>(temb, rtypes, embB);
  launch_gemm(embB,    f1W, f1b,     newfeat, Bsz, Ssz, Ssz, 0, stream);
  launch_gemm(agginit, f2W, nullptr, newfeat, Bsz, Asz, Ssz, 1, stream);  // +=

  // assemble output: [nodes ; new_feats], loss already in its slot
  hipMemcpyAsync(out, nodes, (size_t)Nn * Ssz * 4, hipMemcpyDeviceToDevice, stream);
  hipMemcpyAsync(out + (size_t)Nn * Ssz, newfeat, (size_t)Bsz * Ssz * 4,
                 hipMemcpyDeviceToDevice, stream);
}
